// Int8LlamaMLP_30940944401185
// MI455X (gfx1250) — 8-bit matrix operand order repaired and checked by reading, compiled, not run on hardware
//
#include <hip/hip_runtime.h>

typedef int v2i __attribute__((ext_vector_type(2)));
typedef int v4i __attribute__((ext_vector_type(4)));
typedef int v8i __attribute__((ext_vector_type(8)));

__device__ __forceinline__ v8i load_a_frag(const signed char* base_row, int lane) {
  const int hi = (lane >> 4) ? 16 : 0;
  v8i a;
#pragma unroll
  for (int c = 0; c < 4; ++c) {
    v2i d = *reinterpret_cast<const v2i*>(base_row + hi + (c & 1) * 8 + (c >> 1) * 32);
    a[2 * c]     = d[0];
    a[2 * c + 1] = d[1];
  }
  return a;
}

__device__ __forceinline__ v8i load_b_frag(const signed char* wrow, int lane) {
  const int hi = (lane >> 4) ? 16 : 0;
  v4i lo = *reinterpret_cast<const v4i*>(wrow + hi);
  v4i up = *reinterpret_cast<const v4i*>(wrow + hi + 32);
  v8i b;
  b[0] = lo[0]; b[1] = lo[1]; b[2] = lo[2]; b[3] = lo[3];
  b[4] = up[0]; b[5] = up[1]; b[6] = up[2]; b[7] = up[3];
  return b;
}

template <int H>
__global__ __launch_bounds__(256) void quant_x_kernel(
    const float* __restrict__ x, signed char* __restrict__ qx,
    float* __restrict__ sx) {
  constexpr int SWEEPS = H / (256 * 4);
  __shared__ float smax[256];
  const int token = blockIdx.x;
  const int tid = threadIdx.x;
  const float4* xt = reinterpret_cast<const float4*>(x + (size_t)token * H);

  float4 v[SWEEPS];
  float m = 0.f;
#pragma unroll
  for (int s = 0; s < SWEEPS; ++s) {
    v[s] = xt[s * 256 + tid];
    m = fmaxf(m, fmaxf(fmaxf(fabsf(v[s].x), fabsf(v[s].y)),
                       fmaxf(fabsf(v[s].z), fabsf(v[s].w))));
  }
  smax[tid] = m;
  __syncthreads();
#pragma unroll
  for (int off = 128; off > 0; off >>= 1) {
    if (tid < off) smax[tid] = fmaxf(smax[tid], smax[tid + off]);
    __syncthreads();
  }
  const float scale = fmaxf(smax[0] * (1.f / 127.f), 1e-8f);
  const float inv = 1.f / scale;

  int* qout = reinterpret_cast<int*>(qx + (size_t)token * H);
#pragma unroll
  for (int s = 0; s < SWEEPS; ++s) {
    int b0 = (int)rintf(fminf(fmaxf(v[s].x * inv, -127.f), 127.f));
    int b1 = (int)rintf(fminf(fmaxf(v[s].y * inv, -127.f), 127.f));
    int b2 = (int)rintf(fminf(fmaxf(v[s].z * inv, -127.f), 127.f));
    int b3 = (int)rintf(fminf(fmaxf(v[s].w * inv, -127.f), 127.f));
    qout[s * 256 + tid] =
        (b0 & 255) | ((b1 & 255) << 8) | ((b2 & 255) << 16) | ((b3 & 255) << 24);
  }
  if (tid == 0) sx[token] = scale;
}

__global__ __launch_bounds__(256) void gateup_kernel(
    const signed char* __restrict__ qx, const float* __restrict__ sx,
    const signed char* __restrict__ wg, const signed char* __restrict__ wu,
    const float* __restrict__ gscale, const float* __restrict__ uscale,
    const float* __restrict__ dis_ptr, signed char* __restrict__ qh,
    int M, int H, int I) {
  const int lane = threadIdx.x & 31;
  const int wave = threadIdx.x >> 5;
  const int wm = wave & 1;
  const int wn = wave >> 1;
  const int rowBase = blockIdx.y * 128 + wm * 64;
  const int colBase = blockIdx.x * 128 + wn * 32;
  if (rowBase >= M) return;

  const int arow = lane & 15;

  const v8i zero = {0, 0, 0, 0, 0, 0, 0, 0};
  v8i accg[4][2], accu[4][2];
#pragma unroll
  for (int t = 0; t < 4; ++t)
#pragma unroll
    for (int n = 0; n < 2; ++n) { accg[t][n] = zero; accu[t][n] = zero; }

  const signed char* aptr[4];
#pragma unroll
  for (int t = 0; t < 4; ++t)
    aptr[t] = qx + (size_t)(rowBase + t * 16 + arow) * H;
  const signed char* gptr[2];
  const signed char* uptr[2];
#pragma unroll
  for (int n = 0; n < 2; ++n) {
    const int bcol = colBase + n * 16 + (lane & 15);
    gptr[n] = wg + (size_t)bcol * H;
    uptr[n] = wu + (size_t)bcol * H;
  }

  for (int k0 = 0; k0 < H; k0 += 64) {
    v8i bg[2], bu[2];
#pragma unroll
    for (int n = 0; n < 2; ++n) {
      bg[n] = load_b_frag(gptr[n] + k0, lane);
      bu[n] = load_b_frag(uptr[n] + k0, lane);
    }
#pragma unroll
    for (int t = 0; t < 4; ++t) {
      const v8i a = load_a_frag(aptr[t] + k0, lane);
#pragma unroll
      for (int n = 0; n < 2; ++n) {
        accg[t][n] = __builtin_amdgcn_wmma_i32_16x16x64_iu8(
            true, a, true, bg[n], accg[t][n], false, false);
        accu[t][n] = __builtin_amdgcn_wmma_i32_16x16x64_iu8(
            true, a, true, bu[n], accu[t][n], false, false);
      }
    }
  }

  const float inv_dis = 1.f / dis_ptr[0];
  const int sel = lane >> 4;
#pragma unroll
  for (int n = 0; n < 2; ++n) {
    const int bcol = colBase + n * 16 + (lane & 15);
    const float gsc = gscale[bcol];
    const float usc = uscale[bcol];
#pragma unroll
    for (int t = 0; t < 4; ++t) {
#pragma unroll
      for (int r = 0; r < 8; ++r) {
        const int row = rowBase + t * 16 + sel * 8 + r;
        const float sxv = sx[row];
        const float g = (float)accg[t][n][r] * sxv * gsc;
        const float u = (float)accu[t][n][r] * sxv * usc;
        const float h = g * (1.f / (1.f + __expf(-g))) * u;
        const int q = (int)rintf(fminf(fmaxf(h * inv_dis, -127.f), 127.f));
        qh[(size_t)row * I + bcol] = (signed char)q;
      }
    }
  }
}

__global__ __launch_bounds__(256) void down_kernel(
    const signed char* __restrict__ qh, const signed char* __restrict__ wd,
    const float* __restrict__ dscale, const float* __restrict__ dis_ptr,
    float* __restrict__ out, int M, int I, int H) {
  const int lane = threadIdx.x & 31;
  const int wave = threadIdx.x >> 5;
  const int wm = wave & 1;
  const int wn = wave >> 1;
  const int rowBase = blockIdx.y * 128 + wm * 64;
  const int colBase = blockIdx.x * 128 + wn * 32;
  if (rowBase >= M) return;

  const int arow = lane & 15;

  const v8i zero = {0, 0, 0, 0, 0, 0, 0, 0};
  v8i acc[4][2];
#pragma unroll
  for (int t = 0; t < 4; ++t)
#pragma unroll
    for (int n = 0; n < 2; ++n) acc[t][n] = zero;

  const signed char* aptr[4];
#pragma unroll
  for (int t = 0; t < 4; ++t)
    aptr[t] = qh + (size_t)(rowBase + t * 16 + arow) * I;
  const signed char* bptr[2];
#pragma unroll
  for (int n = 0; n < 2; ++n)
    bptr[n] = wd + (size_t)(colBase + n * 16 + (lane & 15)) * I;

  for (int k0 = 0; k0 < I; k0 += 64) {
    v8i b[2];
#pragma unroll
    for (int n = 0; n < 2; ++n) b[n] = load_b_frag(bptr[n] + k0, lane);
#pragma unroll
    for (int t = 0; t < 4; ++t) {
      const v8i a = load_a_frag(aptr[t] + k0, lane);
#pragma unroll
      for (int n = 0; n < 2; ++n)
        acc[t][n] = __builtin_amdgcn_wmma_i32_16x16x64_iu8(
            true, a, true, b[n], acc[t][n], false, false);
    }
  }

  const float dis = dis_ptr[0];
  const int sel = lane >> 4;
#pragma unroll
  for (int n = 0; n < 2; ++n) {
    const int bcol = colBase + n * 16 + (lane & 15);
    const float dsc = dis * dscale[bcol];
#pragma unroll
    for (int t = 0; t < 4; ++t) {
#pragma unroll
      for (int r = 0; r < 8; ++r) {
        const int row = rowBase + t * 16 + sel * 8 + r;
        out[(size_t)row * H + bcol] = (float)acc[t][n][r] * dsc;
      }
    }
  }
}

extern "C" void kernel_launch(void* const* d_in, const int* in_sizes, int n_in,
                              void* d_out, int out_size, void* d_ws,
                              size_t ws_size, hipStream_t stream) {
  const float* x             = (const float*)d_in[0];
  const signed char* w_gate  = (const signed char*)d_in[1];
  const signed char* w_up    = (const signed char*)d_in[2];
  const signed char* w_down  = (const signed char*)d_in[3];
  const float* gate_scale    = (const float*)d_in[4];
  const float* up_scale      = (const float*)d_in[5];
  const float* down_scale    = (const float*)d_in[6];
  const float* down_in_scale = (const float*)d_in[7];

  const int I = in_sizes[4];
  const int H = in_sizes[6];
  const int M = in_sizes[0] / H;

  char* ws = (char*)d_ws;
  size_t off = 0;
  signed char* qx = (signed char*)(ws + off);
  off += ((size_t)M * H + 255) & ~(size_t)255;
  float* sx = (float*)(ws + off);
  off += ((size_t)M * sizeof(float) + 255) & ~(size_t)255;
  signed char* qh = (signed char*)(ws + off);

  quant_x_kernel<4096><<<M, 256, 0, stream>>>(x, qx, sx);

  dim3 g1(I / 128, M / 128);
  gateup_kernel<<<g1, 256, 0, stream>>>(qx, sx, w_gate, w_up, gate_scale,
                                        up_scale, down_in_scale, qh, M, H, I);

  dim3 g2(H / 128, M / 128);
  down_kernel<<<g2, 256, 0, stream>>>(qh, w_down, down_scale, down_in_scale,
                                      (float*)d_out, M, I, H);
}
